// EGRU_50422916055692
// MI455X (gfx1250) — compile-verified, matrix operand order checked by reading, not run on hardware
//
#include <hip/hip_runtime.h>

// ---------------------------------------------------------------------------
// EGRU forward for MI455X (gfx1250, wave32, WMMA 16x16x32 bf16, f32 accum)
//
// Shapes: T=256, B=128, I=1024, H=1536, 3H=4608
// Per-step fused kernel: one wave owns a 16(B)x16(H) tile with 4 accumulators
// (z, r, gx, gh), runs K over I (input proj) then H (recurrent proj), then
// applies gates + spike + trace and writes y/h/o/trace for t+1.
// ---------------------------------------------------------------------------

#define T_ 256
#define B_ 128
#define I_ 1024
#define H_ 1536
#define N3_ 4608

typedef __attribute__((ext_vector_type(8)))  __bf16 v8bf;
typedef __attribute__((ext_vector_type(16))) __bf16 v16bf;
typedef __attribute__((ext_vector_type(8)))  float  v8f;

__device__ __forceinline__ v16bf load_frag(const __bf16* __restrict__ p) {
    // per-lane: 8 contiguous bf16 at p (K k..k+7) and 8 at p+16 (K k+16..k+23)
    v8bf lo = *reinterpret_cast<const v8bf*>(p);
    v8bf hi = *reinterpret_cast<const v8bf*>(p + 16);
    v16bf r;
#pragma unroll
    for (int i = 0; i < 8; ++i) { r[i] = lo[i]; r[i + 8] = hi[i]; }
    return r;
}

__device__ __forceinline__ v8f wmma_bf16(v16bf a, v16bf b, v8f c) {
    return __builtin_amdgcn_wmma_f32_16x16x32_bf16(
        /*neg_a=*/false, a, /*neg_b=*/false, b,
        /*c_mod=*/(short)0, c, /*reuse_a=*/false, /*reuse_b=*/false);
}

__device__ __forceinline__ float sigmoidf_(float x) {
    return 1.0f / (1.0f + __expf(-x));
}

// ---------------------------------------------------------------------------
// One-time prep kernels
// ---------------------------------------------------------------------------

__global__ __launch_bounds__(256) void cvt_f32_to_bf16(
    const float* __restrict__ src, __bf16* __restrict__ dst, long n) {
    long i = (long)blockIdx.x * blockDim.x + threadIdx.x;
    if (i < n) dst[i] = (__bf16)src[i];
}

// dst[n*K + k] = (bf16) src[k*N + n]   (src is [K,N] row-major -> dst [N,K])
__global__ __launch_bounds__(256) void transpose_cvt(
    const float* __restrict__ src, __bf16* __restrict__ dst, int K, int N) {
    long idx = (long)blockIdx.x * blockDim.x + threadIdx.x;
    if (idx >= (long)K * N) return;
    int n = (int)(idx / K);
    int k = (int)(idx % K);
    dst[idx] = (__bf16)src[(long)k * N + n];
}

// init state + t=0 output slabs (y[0]=h0, h[0]=o[0]=trace[0]=0)
__global__ __launch_bounds__(256) void egru_init(
    const float* __restrict__ h0, __bf16* __restrict__ y_cur,
    float* __restrict__ h_state, float* __restrict__ trace_state,
    float* __restrict__ out) {
    int i = blockIdx.x * blockDim.x + threadIdx.x;
    if (i >= B_ * H_) return;
    const size_t T1BH = (size_t)(T_ + 1) * B_ * H_;
    float y0 = h0[i];
    y_cur[i]       = (__bf16)y0;
    h_state[i]     = 0.0f;
    trace_state[i] = 0.0f;
    out[i]            = y0;    // y[0]
    out[T1BH + i]     = 0.0f;  // h[0]
    out[2 * T1BH + i] = 0.0f;  // o[0]
    out[3 * T1BH + i] = 0.0f;  // trace[0]
}

// ---------------------------------------------------------------------------
// Fused per-timestep kernel: dual GEMM (input proj + recurrent proj) + gates
// grid = (96/4, 8) blocks of 128 threads; wave w in block owns N-tile
// ---------------------------------------------------------------------------

__global__ __launch_bounds__(128) void egru_step(
    const __bf16* __restrict__ xbf,      // [T,B,I] bf16
    const __bf16* __restrict__ y_cur,    // [B,H] bf16
    const __bf16* __restrict__ Wt,       // [3H,I] bf16 (transposed kernel)
    const __bf16* __restrict__ Rt,       // [3H,H] bf16 (transposed recurrent)
    const float*  __restrict__ bias,     // [3H]
    const float*  __restrict__ rbias,    // [3H]
    const float*  __restrict__ thr,      // [H]
    float*        __restrict__ h_state,  // [B,H]
    float*        __restrict__ trace_state, // [B,H]
    __bf16*       __restrict__ y_next,   // [B,H] bf16
    float*        __restrict__ out,      // [4, T+1, B, H]
    int t) {
    const int wave  = threadIdx.x >> 5;
    const int lane  = threadIdx.x & 31;
    const int half  = lane >> 4;       // K-half select per WMMA A/B layout
    const int l16   = lane & 15;
    const int tileN = blockIdx.x * 4 + wave;  // 0..95
    const int tileM = blockIdx.y;             // 0..7
    const int n0 = tileN * 16;
    const int m0 = tileM * 16;
    const int arow = m0 + l16;         // batch row this lane stripes (A operand)
    const int koff = half * 8;

    v8f acc_z  = {};
    v8f acc_r  = {};
    v8f acc_gx = {};
    v8f acc_gh = {};

    // ---- phase 1: input projection, K over I ----
    {
        const __bf16* __restrict__ xr = xbf + ((size_t)t * B_ + arow) * I_ + koff;
        const __bf16* __restrict__ wz = Wt + (size_t)(n0 + l16) * I_ + koff;
        const __bf16* __restrict__ wr = Wt + (size_t)(H_ + n0 + l16) * I_ + koff;
        const __bf16* __restrict__ wg = Wt + (size_t)(2 * H_ + n0 + l16) * I_ + koff;
#pragma unroll 4
        for (int k = 0; k < I_; k += 32) {
            v16bf a  = load_frag(xr + k);
            v16bf bz = load_frag(wz + k);
            v16bf br = load_frag(wr + k);
            v16bf bg = load_frag(wg + k);
            acc_z  = wmma_bf16(a, bz, acc_z);
            acc_r  = wmma_bf16(a, br, acc_r);
            acc_gx = wmma_bf16(a, bg, acc_gx);
        }
    }
    // ---- phase 2: recurrent projection, K over H ----
    {
        const __bf16* __restrict__ yr = y_cur + (size_t)arow * H_ + koff;
        const __bf16* __restrict__ rz = Rt + (size_t)(n0 + l16) * H_ + koff;
        const __bf16* __restrict__ rr = Rt + (size_t)(H_ + n0 + l16) * H_ + koff;
        const __bf16* __restrict__ rg = Rt + (size_t)(2 * H_ + n0 + l16) * H_ + koff;
#pragma unroll 4
        for (int k = 0; k < H_; k += 32) {
            v16bf a  = load_frag(yr + k);
            v16bf bz = load_frag(rz + k);
            v16bf br = load_frag(rr + k);
            v16bf bg = load_frag(rg + k);
            acc_z  = wmma_bf16(a, bz, acc_z);
            acc_r  = wmma_bf16(a, br, acc_r);
            acc_gh = wmma_bf16(a, bg, acc_gh);
        }
    }

    // ---- epilogue: gates, spike, trace, outputs ----
    // C/D layout: VGPR j, lanes 0-15 -> M=j, lanes 16-31 -> M=j+8; N = lane&15
    const int gcol = n0 + l16;                 // hidden index 0..H-1
    const float bz  = bias[gcol]          + rbias[gcol];
    const float brr = bias[H_ + gcol]     + rbias[H_ + gcol];
    const float bgx = bias[2 * H_ + gcol];
    const float bgh = rbias[2 * H_ + gcol];
    const float th  = thr[gcol];
    const size_t T1BH = (size_t)(T_ + 1) * B_ * H_;
    const size_t tbase = (size_t)(t + 1) * B_ * H_;

#pragma unroll
    for (int j = 0; j < 8; ++j) {
        const int m = m0 + j + 8 * half;
        const size_t idx = (size_t)m * H_ + gcol;
        float z = sigmoidf_(acc_z[j] + bz);
        float r = sigmoidf_(acc_r[j] + brr);
        float g = tanhf(acc_gx[j] + bgx + r * (acc_gh[j] + bgh));
        float h = h_state[idx];
        float cur = z * h + (1.0f - z) * g;
        float ev  = (cur > th) ? 1.0f : 0.0f;
        float nh  = cur - ev * th;
        float ny  = ev * cur;
        h_state[idx] = nh;
        y_next[idx]  = (__bf16)ny;
        float tr = 0.9f * trace_state[idx] + 0.1f * ny;
        trace_state[idx] = tr;
        const size_t o = tbase + idx;
        out[o]            = ny;  // y
        out[T1BH + o]     = nh;  // h
        out[2 * T1BH + o] = ev;  // o (event)
        out[3 * T1BH + o] = tr;  // trace
    }
}

// ---------------------------------------------------------------------------
// Host launcher
// ---------------------------------------------------------------------------

extern "C" void kernel_launch(void* const* d_in, const int* in_sizes, int n_in,
                              void* d_out, int out_size, void* d_ws, size_t ws_size,
                              hipStream_t stream) {
    const float* input   = (const float*)d_in[0];  // [T,B,I]
    const float* h0      = (const float*)d_in[1];  // [B,H]
    const float* kernel_ = (const float*)d_in[2];  // [I,3H]
    const float* rkernel = (const float*)d_in[3];  // [H,3H]
    const float* bias    = (const float*)d_in[4];  // [3H]
    const float* rbias   = (const float*)d_in[5];  // [3H]
    const float* thr     = (const float*)d_in[6];  // [H]
    float* out = (float*)d_out;

    // workspace layout (all 256B-aligned)
    char* ws = (char*)d_ws;
    size_t off = 0;
    auto alloc = [&](size_t bytes) {
        char* p = ws + off;
        off += (bytes + 255) & ~(size_t)255;
        return p;
    };
    __bf16* xbf    = (__bf16*)alloc((size_t)T_ * B_ * I_ * 2);   // 64 MiB
    __bf16* Wt     = (__bf16*)alloc((size_t)N3_ * I_ * 2);       //  9 MiB
    __bf16* Rt     = (__bf16*)alloc((size_t)N3_ * H_ * 2);       // 13.5 MiB
    __bf16* yping  = (__bf16*)alloc((size_t)B_ * H_ * 2);
    __bf16* ypong  = (__bf16*)alloc((size_t)B_ * H_ * 2);
    float*  hstate = (float*)alloc((size_t)B_ * H_ * 4);
    float*  tstate = (float*)alloc((size_t)B_ * H_ * 4);
    (void)ws_size; (void)in_sizes; (void)n_in; (void)out_size;

    // --- one-time prep ---
    {
        long n = (long)T_ * B_ * I_;
        cvt_f32_to_bf16<<<(unsigned)((n + 255) / 256), 256, 0, stream>>>(input, xbf, n);
    }
    {
        long n = (long)N3_ * I_;
        transpose_cvt<<<(unsigned)((n + 255) / 256), 256, 0, stream>>>(kernel_, Wt, I_, N3_);
    }
    {
        long n = (long)N3_ * H_;
        transpose_cvt<<<(unsigned)((n + 255) / 256), 256, 0, stream>>>(rkernel, Rt, H_, N3_);
    }
    egru_init<<<(B_ * H_ + 255) / 256, 256, 0, stream>>>(h0, yping, hstate, tstate, out);

    // --- sequential recurrence: one fused kernel per timestep ---
    dim3 grid(96 / 4, 8);  // 96 N-tiles (4 per block), 8 M-tiles
    for (int t = 0; t < T_; ++t) {
        __bf16* ycur  = (t & 1) ? ypong : yping;
        __bf16* ynext = (t & 1) ? yping : ypong;
        egru_step<<<grid, 128, 0, stream>>>(xbf, ycur, Wt, Rt, bias, rbias, thr,
                                            hstate, tstate, ynext, out, t);
    }
}
